// HGNN_85186381349629
// MI455X (gfx1250) — matrix operand order repaired and checked by reading, not compiled, not run on hardware
//
#include <hip/hip_runtime.h>
#include <hip/hip_bf16.h>
#include <math.h>

// ---------------------------------------------------------------------------
// HGNN forward for MI455X (gfx1250, wave32, WMMA).
// fp32 in global memory; GEMMs stage tiles to LDS with f32->bf16 conversion
// (v_cvt_pk_bf16_f32) and run v_wmma_f32_16x16x32_bf16 with fp32 accumulate.
// Staging is register-pipelined: next K-tile is loaded (float4/b128) while the
// current tile's 8 WMMAs execute, so global latency hides behind math.
// ---------------------------------------------------------------------------

#define BATCH 2
#define NDIM 2048
#define CDIM 128
#define INCH 6

typedef __attribute__((ext_vector_type(16))) __bf16 v16bf;
typedef __attribute__((ext_vector_type(4)))  __bf16 v4bf;
typedef __attribute__((ext_vector_type(8)))  float  v8f;
typedef __bf16 bf16_t;

#define TBM 128
#define TBN 128
#define TBK 32
#define LDSTR 40   // padded LDS row stride (bf16): 80B rows -> aligned, low-conflict

// D = alpha * op(A) . op(B) [+ bias[m]] [+ resid]   (logical A: MxK, B: KxN)
// TA: A_phys is KxM row-major (A[m,k] = Ap[k*lda+m]); TB: B_phys is NxK.
// Requirements (met by all call sites): K % 32 == 0, lda/ldb multiples of 4,
// 16B-aligned base pointers.
template<bool TA, bool TB>
__global__ void __launch_bounds__(256)
gemm_bf16_wmma(const float* __restrict__ A, const float* __restrict__ B,
               float* __restrict__ D, int M, int N, int K,
               int lda, int ldb, int ldd,
               long long sA, long long sB, long long sD,
               float alpha, const float* __restrict__ bias,
               const float* __restrict__ resid)
{
    __shared__ bf16_t As[TBM][LDSTR];
    __shared__ bf16_t Bs[TBN][LDSTR];

    const int tid  = threadIdx.x;
    const int wave = tid >> 5;
    const int lane = tid & 31;
    const int wm = wave >> 2;          // 0..1 : 64 rows each
    const int wn = wave & 3;           // 0..3 : 32 cols each
    const int lh = lane & 15;
    const int hf = lane >> 4;          // lane half
    const int m0 = blockIdx.y * TBM;
    const int n0 = blockIdx.x * TBN;
    const int bz = blockIdx.z;

    A += (long long)bz * sA;
    B += (long long)bz * sB;
    D += (long long)bz * sD;
    const float* R = resid ? resid + (long long)bz * sD : (const float*)0;

    const bool fullM = (m0 + TBM <= M);
    const bool fullN = (n0 + TBN <= N);

    // staging thread maps (4x float4 per thread per operand)
    const int tA_k  = tid >> 5;         // TA : k row (0..7, +p*8)
    const int tA_m  = (tid & 31) << 2;  // TA : m col group
    const int tA_r  = tid >> 3;         // !TA: m row (0..31, +p*32)
    const int tA_kg = (tid & 7) << 2;   // !TA: k col group
    const int tB_n  = tid >> 3;         // TB : n row (0..31, +p*32)
    const int tB_kg = (tid & 7) << 2;   // TB : k col group
    const int tB_k  = tid >> 5;         // !TB: k row (0..7, +p*8)
    const int tB_ng = (tid & 31) << 2;  // !TB: n col group

    float4 ra[4], rb[4];

    auto loadTiles = [&](int kpos) {
        if (TA) {
#pragma unroll
            for (int p = 0; p < 4; ++p) {
                long long off = (long long)(kpos + tA_k + p * 8) * lda + m0 + tA_m;
                if (fullM) {
                    ra[p] = *(const float4*)(A + off);
                } else {
                    float4 v; v.x = v.y = v.z = v.w = 0.f;
                    int mb = m0 + tA_m;
                    if (mb + 0 < M) v.x = A[off + 0];
                    if (mb + 1 < M) v.y = A[off + 1];
                    if (mb + 2 < M) v.z = A[off + 2];
                    if (mb + 3 < M) v.w = A[off + 3];
                    ra[p] = v;
                }
            }
        } else {
#pragma unroll
            for (int p = 0; p < 4; ++p) {
                int gm = m0 + tA_r + p * 32;
                long long off = (long long)gm * lda + kpos + tA_kg;
                if (gm < M) ra[p] = *(const float4*)(A + off);
                else { float4 v; v.x = v.y = v.z = v.w = 0.f; ra[p] = v; }
            }
        }
        if (TB) {
#pragma unroll
            for (int p = 0; p < 4; ++p) {
                int gn = n0 + tB_n + p * 32;
                long long off = (long long)gn * ldb + kpos + tB_kg;
                if (gn < N) rb[p] = *(const float4*)(B + off);
                else { float4 v; v.x = v.y = v.z = v.w = 0.f; rb[p] = v; }
            }
        } else {
#pragma unroll
            for (int p = 0; p < 4; ++p) {
                long long off = (long long)(kpos + tB_k + p * 8) * ldb + n0 + tB_ng;
                if (fullN) {
                    rb[p] = *(const float4*)(B + off);
                } else {
                    float4 v; v.x = v.y = v.z = v.w = 0.f;
                    int nb = n0 + tB_ng;
                    if (nb + 0 < N) v.x = B[off + 0];
                    if (nb + 1 < N) v.y = B[off + 1];
                    if (nb + 2 < N) v.z = B[off + 2];
                    if (nb + 3 < N) v.w = B[off + 3];
                    rb[p] = v;
                }
            }
        }
    };

    auto storeTiles = [&]() {
        if (TA) {  // transpose in LDS: scalar b16 stores
#pragma unroll
            for (int p = 0; p < 4; ++p) {
                int kq = tA_k + p * 8;
                As[tA_m + 0][kq] = (bf16_t)ra[p].x;
                As[tA_m + 1][kq] = (bf16_t)ra[p].y;
                As[tA_m + 2][kq] = (bf16_t)ra[p].z;
                As[tA_m + 3][kq] = (bf16_t)ra[p].w;
            }
        } else {   // packed 4x bf16 (8B) stores
#pragma unroll
            for (int p = 0; p < 4; ++p) {
                v4bf t;
                t[0] = (bf16_t)ra[p].x; t[1] = (bf16_t)ra[p].y;
                t[2] = (bf16_t)ra[p].z; t[3] = (bf16_t)ra[p].w;
                *(v4bf*)&As[tA_r + p * 32][tA_kg] = t;
            }
        }
        if (TB) {
#pragma unroll
            for (int p = 0; p < 4; ++p) {
                v4bf t;
                t[0] = (bf16_t)rb[p].x; t[1] = (bf16_t)rb[p].y;
                t[2] = (bf16_t)rb[p].z; t[3] = (bf16_t)rb[p].w;
                *(v4bf*)&Bs[tB_n + p * 32][tB_kg] = t;
            }
        } else {
#pragma unroll
            for (int p = 0; p < 4; ++p) {
                int kq = tB_k + p * 8;
                Bs[tB_ng + 0][kq] = (bf16_t)rb[p].x;
                Bs[tB_ng + 1][kq] = (bf16_t)rb[p].y;
                Bs[tB_ng + 2][kq] = (bf16_t)rb[p].z;
                Bs[tB_ng + 3][kq] = (bf16_t)rb[p].w;
            }
        }
    };

    v8f acc[4][2];
#pragma unroll
    for (int mi = 0; mi < 4; ++mi)
#pragma unroll
        for (int ni = 0; ni < 2; ++ni)
#pragma unroll
            for (int r = 0; r < 8; ++r) acc[mi][ni][r] = 0.f;

    loadTiles(0);

    for (int kk = 0; kk < K; kk += TBK) {
        __syncthreads();
        storeTiles();               // waits for in-flight loads here
        __syncthreads();

        if (kk + TBK < K) {         // issue next tile's loads; waited next iter
            loadTiles(kk + TBK);
            if (kk + 2 * TBK < K) { // warm L2 for the tile after next
                __builtin_prefetch(TA ? &A[(long long)(kk + 2 * TBK) * lda + m0]
                                      : &A[(long long)m0 * lda + (kk + 2 * TBK)], 0, 0);
                __builtin_prefetch(TB ? &B[(long long)n0 * ldb + (kk + 2 * TBK)]
                                      : &B[(long long)(kk + 2 * TBK) * ldb + n0], 0, 0);
            }
        }

        // B fragments: lane = one column; lanes 0-15: K 0-7/16-23, 16-31: 8-15/24-31
        v16bf bfrag[2];
#pragma unroll
        for (int ni = 0; ni < 2; ++ni) {
            int col = wn * 32 + ni * 16 + lh;
            int kb  = hf * 8;
#pragma unroll
            for (int e = 0; e < 16; ++e) bfrag[ni][e] = Bs[col][kb + e + (e >= 8 ? 8 : 0)];
        }
#pragma unroll
        for (int mi = 0; mi < 4; ++mi) {
            int row = wm * 64 + mi * 16 + lh;
            int ka  = hf * 8;       // lanes 0-15: K 0-7/16-23, 16-31: 8-15/24-31
            v16bf afrag;
#pragma unroll
            for (int j = 0; j < 8; ++j) {
                afrag[j]     = As[row][ka + j];
                afrag[8 + j] = As[row][ka + 16 + j];
            }
#pragma unroll
            for (int ni = 0; ni < 2; ++ni) {
                acc[mi][ni] = __builtin_amdgcn_wmma_f32_16x16x32_bf16(
                    false, afrag, false, bfrag[ni], (short)0, acc[mi][ni], false, false);
            }
        }
    }

    // ---- epilogue: D/C layout (VGPR r -> row r / r+8 per lane half) ----
    if (fullM && fullN) {
#pragma unroll
        for (int mi = 0; mi < 4; ++mi)
#pragma unroll
            for (int ni = 0; ni < 2; ++ni)
#pragma unroll
                for (int r = 0; r < 8; ++r) {
                    int gm = m0 + wm * 64 + mi * 16 + (hf ? r + 8 : r);
                    int gn = n0 + wn * 32 + ni * 16 + lh;
                    float v = alpha * acc[mi][ni][r];
                    if (bias) v += bias[gm];
                    long long o = (long long)gm * ldd + gn;
                    if (R) v += R[o];
                    D[o] = v;
                }
    } else {
#pragma unroll
        for (int mi = 0; mi < 4; ++mi)
#pragma unroll
            for (int ni = 0; ni < 2; ++ni)
#pragma unroll
                for (int r = 0; r < 8; ++r) {
                    int gm = m0 + wm * 64 + mi * 16 + (hf ? r + 8 : r);
                    int gn = n0 + wn * 32 + ni * 16 + lh;
                    if (gm < M && gn < N) {
                        float v = alpha * acc[mi][ni][r];
                        if (bias) v += bias[gm];
                        long long o = (long long)gm * ldd + gn;
                        if (R) v += R[o];
                        D[o] = v;
                    }
                }
    }
}

// ---------------------------------------------------------------------------
// Support kernels
// ---------------------------------------------------------------------------

__global__ void k_binarize(const float* __restrict__ W, float* __restrict__ H,
                           float* __restrict__ rawH, float* __restrict__ deg)
{
    int b = blockIdx.z;
    int e = blockIdx.x * blockDim.x + threadIdx.x;
    if (e >= NDIM) return;
    long long base = (long long)b * NDIM * NDIM;
    float s = 0.f;
    for (int v = 0; v < NDIM; ++v) {
        long long o = base + (long long)v * NDIM + e;
        float h = (W[o] > 0.f) ? 1.f : 0.f;
        H[o] = h; rawH[o] = h; s += h;
    }
    deg[b * NDIM + e] = s;
}

__global__ void k_colsum(const float* __restrict__ Mt, float* __restrict__ out)
{
    int b = blockIdx.z;
    int e = blockIdx.x * blockDim.x + threadIdx.x;
    if (e >= NDIM) return;
    long long base = (long long)b * NDIM * NDIM;
    float s = 0.f;
    for (int v = 0; v < NDIM; ++v) s += Mt[base + (long long)v * NDIM + e];
    out[b * NDIM + e] = s;
}

__global__ void k_rowsum(const float* __restrict__ Mt, float* __restrict__ out)
{
    __shared__ float red[256];
    int b = blockIdx.z, v = blockIdx.x;
    long long base = ((long long)b * NDIM + v) * NDIM;
    float s = 0.f;
    for (int i = threadIdx.x; i < NDIM; i += 256) s += Mt[base + i];
    red[threadIdx.x] = s;
    for (int st = 128; st > 0; st >>= 1) {
        __syncthreads();
        if (threadIdx.x < st) red[threadIdx.x] += red[threadIdx.x + st];
    }
    if (threadIdx.x == 0) out[b * NDIM + v] = red[0];
}

__global__ void k_inv(const float* __restrict__ x, float* __restrict__ out, int n)
{
    int i = blockIdx.x * blockDim.x + threadIdx.x;
    if (i < n) { float v = x[i]; out[i] = (v > 0.f) ? 1.f / v : 0.f; }
}

__global__ void k_l2norm(const float* __restrict__ x, float* __restrict__ out)
{
    __shared__ float red[256];
    int b = blockIdx.x;
    float s = 0.f;
    for (int i = threadIdx.x; i < NDIM; i += 256) { float v = x[b * NDIM + i]; s += v * v; }
    red[threadIdx.x] = s;
    for (int st = 128; st > 0; st >>= 1) {
        __syncthreads();
        if (threadIdx.x < st) red[threadIdx.x] += red[threadIdx.x + st];
    }
    __syncthreads();
    float nrm = fmaxf(sqrtf(red[0]), 1e-12f);
    for (int i = threadIdx.x; i < NDIM; i += 256) out[b * NDIM + i] = x[b * NDIM + i] / nrm;
}

__global__ void k_conv_in(const float* __restrict__ x, const float* __restrict__ w0,
                          const float* __restrict__ b0, float* __restrict__ xx)
{
    int b = blockIdx.z, c = blockIdx.y;
    int n = blockIdx.x * blockDim.x + threadIdx.x;
    if (n >= NDIM) return;
    float s = b0[c];
#pragma unroll
    for (int j = 0; j < INCH; ++j)
        s += w0[c * INCH + j] * x[((long long)b * INCH + j) * NDIM + n];
    xx[((long long)b * CDIM + c) * NDIM + n] = s;
}

__global__ void k_bnstats(const float* __restrict__ t, float* __restrict__ mean,
                          float* __restrict__ var, int C_)
{
    __shared__ float r1[256], r2[256];
    int c = blockIdx.x;
    float s = 0.f, q = 0.f;
    for (int b = 0; b < BATCH; ++b)
        for (int i = threadIdx.x; i < NDIM; i += 256) {
            float v = t[((long long)b * C_ + c) * NDIM + i];
            s += v; q += v * v;
        }
    r1[threadIdx.x] = s; r2[threadIdx.x] = q;
    for (int st = 128; st > 0; st >>= 1) {
        __syncthreads();
        if (threadIdx.x < st) { r1[threadIdx.x] += r1[threadIdx.x + st]; r2[threadIdx.x] += r2[threadIdx.x + st]; }
    }
    if (threadIdx.x == 0) {
        float m = r1[0] / (float)(BATCH * NDIM);
        mean[c] = m;
        var[c]  = r2[0] / (float)(BATCH * NDIM) - m * m;
    }
}

// act: 0 none, 1 lrelu(0.2), 2 relu
__global__ void k_bnapply(const float* __restrict__ t, const float* __restrict__ mean,
                          const float* __restrict__ var, const float* __restrict__ g,
                          const float* __restrict__ bb, const float* __restrict__ resid,
                          float* __restrict__ out, int C_, int act)
{
    long long idx = (long long)blockIdx.x * blockDim.x + threadIdx.x;
    long long total = (long long)BATCH * C_ * NDIM;
    if (idx >= total) return;
    int c = (int)((idx / NDIM) % C_);
    float v = (t[idx] - mean[c]) * rsqrtf(var[c] + 1e-5f) * g[c] + bb[c];
    if (resid) v += resid[idx];
    if (act == 1) v = v > 0.f ? v : 0.2f * v;
    else if (act == 2) v = fmaxf(v, 0.f);
    out[idx] = v;
}

// out[b,n,c] = scale[b,n] * (transposeSrc ? src[b,c,n] : src[b,n,c])
__global__ void k_rowscale(const float* __restrict__ src, const float* __restrict__ scale,
                           float* __restrict__ out, int Cc, int transposeSrc)
{
    long long idx = (long long)blockIdx.x * blockDim.x + threadIdx.x;
    long long total = (long long)BATCH * NDIM * Cc;
    if (idx >= total) return;
    int c = (int)(idx % Cc);
    long long bn = idx / Cc;
    int n = (int)(bn % NDIM);
    int b = (int)(bn / NDIM);
    float v = transposeSrc ? src[((long long)b * Cc + c) * NDIM + n] : src[idx];
    float sc = scale ? scale[b * NDIM + n] : 1.f;
    out[idx] = sc * v;
}

__global__ void k_concat(const float* __restrict__ y_nc, const float* __restrict__ e_nc,
                         float* __restrict__ cat)
{
    long long idx = (long long)blockIdx.x * blockDim.x + threadIdx.x;
    long long total = (long long)BATCH * NDIM * 2 * CDIM;
    if (idx >= total) return;
    int k = (int)(idx % (2 * CDIM));
    long long bn = idx / (2 * CDIM);
    cat[idx] = (k < CDIM) ? y_nc[bn * CDIM + k] : e_nc[bn * CDIM + (k - CDIM)];
}

__global__ void k_sigmask(float* __restrict__ s, const float* __restrict__ H0)
{
    long long idx = (long long)blockIdx.x * blockDim.x + threadIdx.x;
    long long total = (long long)BATCH * NDIM * NDIM;
    if (idx >= total) return;
    float v = s[idx];
    s[idx] = (H0[idx] > 0.f) ? 1.f / (1.f + expf(-v)) : 0.f;
}

__global__ void k_softmax(float* __restrict__ fa, const float* __restrict__ attw,
                          const float* __restrict__ Hm)
{
    __shared__ float row[NDIM];
    __shared__ float red[256];
    int b = blockIdx.z, o = blockIdx.x;
    long long base = ((long long)b * NDIM + o) * NDIM;
    float mx = -3.0e38f;
    for (int i = threadIdx.x; i < NDIM; i += 256) {
        float a = attw[base + i] * fa[base + i] + (Hm[base + i] > 0.f ? 0.f : -1.0e9f);
        row[i] = a;
        mx = fmaxf(mx, a);
    }
    red[threadIdx.x] = mx;
    for (int st = 128; st > 0; st >>= 1) {
        __syncthreads();
        if (threadIdx.x < st) red[threadIdx.x] = fmaxf(red[threadIdx.x], red[threadIdx.x + st]);
    }
    __syncthreads();
    mx = red[0];
    __syncthreads();
    float s = 0.f;
    for (int i = threadIdx.x; i < NDIM; i += 256) { float e = expf(row[i] - mx); row[i] = e; s += e; }
    red[threadIdx.x] = s;
    for (int st = 128; st > 0; st >>= 1) {
        __syncthreads();
        if (threadIdx.x < st) red[threadIdx.x] += red[threadIdx.x + st];
    }
    __syncthreads();
    float inv = 1.f / red[0];
    for (int i = threadIdx.x; i < NDIM; i += 256) fa[base + i] = row[i] * inv;
}

// exact k-th largest per row via binary search on IEEE bits (scores in [0,1))
__global__ void k_topk(const float* __restrict__ score, const float* __restrict__ rawH,
                       float* __restrict__ Wm, float* __restrict__ Hn, int kSel)
{
    __shared__ float row[NDIM];
    __shared__ int red[256];
    int b = blockIdx.z, o = blockIdx.x;
    long long base = ((long long)b * NDIM + o) * NDIM;
    for (int i = threadIdx.x; i < NDIM; i += 256) row[i] = score[base + i];
    __syncthreads();
    unsigned lo = 0u, hi = 0x3F800001u;     // (0.0, just-above-1.0]
    while (hi - lo > 1u) {
        unsigned mid = lo + ((hi - lo) >> 1);
        float t = __uint_as_float(mid);
        int c = 0;
        for (int i = threadIdx.x; i < NDIM; i += 256) c += (row[i] >= t) ? 1 : 0;
        red[threadIdx.x] = c;
        for (int st = 128; st > 0; st >>= 1) {
            __syncthreads();
            if (threadIdx.x < st) red[threadIdx.x] += red[threadIdx.x + st];
        }
        __syncthreads();
        int total = red[0];
        __syncthreads();
        if (total >= kSel) lo = mid; else hi = mid;
    }
    float t = __uint_as_float(lo);
    for (int i = threadIdx.x; i < NDIM; i += 256) {
        float s = row[i];
        float wm = (s >= t) ? s : 0.f;
        Wm[base + i] = wm;
        Hn[base + i] = (wm > 0.f) ? rawH[base + i] : 0.f;
    }
}

// ---------------------------------------------------------------------------
// Host orchestration
// ---------------------------------------------------------------------------

static inline void launch_gemm(bool ta, bool tb,
    const float* A, const float* Bm, float* D, int M, int N, int K,
    int lda, int ldb, int ldd, long long sA, long long sB, long long sD,
    float alpha, const float* bias, const float* resid, hipStream_t s)
{
    dim3 g((N + TBN - 1) / TBN, (M + TBM - 1) / TBM, BATCH), b(256);
    if (ta) {
        if (tb) gemm_bf16_wmma<true , true ><<<g, b, 0, s>>>(A, Bm, D, M, N, K, lda, ldb, ldd, sA, sB, sD, alpha, bias, resid);
        else    gemm_bf16_wmma<true , false><<<g, b, 0, s>>>(A, Bm, D, M, N, K, lda, ldb, ldd, sA, sB, sD, alpha, bias, resid);
    } else {
        if (tb) gemm_bf16_wmma<false, true ><<<g, b, 0, s>>>(A, Bm, D, M, N, K, lda, ldb, ldd, sA, sB, sD, alpha, bias, resid);
        else    gemm_bf16_wmma<false, false><<<g, b, 0, s>>>(A, Bm, D, M, N, K, lda, ldb, ldd, sA, sB, sD, alpha, bias, resid);
    }
}

extern "C" void kernel_launch(void* const* d_in, const int* in_sizes, int n_in,
                              void* d_out, int out_size, void* d_ws, size_t ws_size,
                              hipStream_t stream)
{
    (void)in_sizes; (void)n_in; (void)out_size; (void)ws_size;
    const float* x  = (const float*)d_in[0];
    const float* Wd = (const float*)d_in[1];
    const float* w0 = (const float*)d_in[2];
    const float* b0 = (const float*)d_in[3];
    // per-layer params, setup_inputs() insertion order (27 tensors per layer)
    auto P = [&](int l, int j) { return (const float*)d_in[4 + l * 27 + j]; };
    enum { CON_W=0, BN_G, BN_B, MLP_W, MLP_B, MLP_G, MLP_BB,
           NL_QW, NL_QB, NL_KW, NL_KB, NL_VW, NL_VB,
           FM1_W, FM1_B, FM1_G, FM1_BB, FM2_W, FM2_B, FM2_G, FM2_BB,
           FM3_W, FM3_B, GU_QW, GU_QB, GU_KW, GU_KB };

    float* ws = (float*)d_ws;
    size_t off = 0;
    auto alloc = [&](size_t n) { float* p = ws + off; off += n; return p; };
    const size_t NN  = (size_t)BATCH * NDIM * NDIM;
    const size_t CN  = (size_t)BATCH * CDIM * NDIM;
    const size_t NC2 = (size_t)BATCH * NDIM * 2 * CDIM;
    const size_t HN  = (size_t)BATCH * (CDIM / 2) * NDIM;
    const size_t VN  = (size_t)BATCH * NDIM;
    const long long NNp  = (long long)NDIM * NDIM;
    const long long CNp  = (long long)CDIM * NDIM;
    const long long NC2p = (long long)NDIM * 2 * CDIM;
    const long long HNp  = (long long)(CDIM / 2) * NDIM;

    float* Hbuf  = alloc(NN);
    float* rawH  = alloc(NN);
    float* Wmbuf = alloc(NN);
    float* fabuf = alloc(NN);
    float* xxA = alloc(CN); float* xxB = alloc(CN);
    float* edgeb[2] = { alloc(CN), alloc(CN) };
    float* featN = alloc(CN);
    float* featW = alloc(CN);
    float* feat2 = alloc(CN);
    float* catb  = alloc(NC2);
    float* Qb = alloc(CN); float* Kb = alloc(CN); float* Vb = alloc(CN);
    float* msg = alloc(CN);
    float* m1 = alloc(HN); float* m2 = alloc(HN);
    float* deg = alloc(VN); float* deinv = alloc(VN); float* dvinv = alloc(VN);
    float* wsum = alloc(VN); float* wedge = alloc(VN); float* dvtmp = alloc(VN);
    float* meanb = alloc(CDIM); float* varb = alloc(CDIM);

    dim3 blk(256);
    auto ewg = [](long long n) { return dim3((unsigned)((n + 255) / 256)); };
    const float SCL = 1.f / sqrtf((float)CDIM);

    // H = (W>0), raw_H, deg = colsum(H); de_inv = dv_inv = 1/deg; W_edge = l2norm(deg)
    k_binarize<<<dim3((NDIM + 255) / 256, 1, BATCH), blk, 0, stream>>>(Wd, Hbuf, rawH, deg);
    k_inv<<<ewg(VN), blk, 0, stream>>>(deg, deinv, (int)VN);
    k_inv<<<ewg(VN), blk, 0, stream>>>(deg, dvinv, (int)VN);
    k_l2norm<<<dim3(BATCH), blk, 0, stream>>>(deg, wedge);
    // xx = w0 @ x + b0
    k_conv_in<<<dim3((NDIM + 255) / 256, CDIM, BATCH), blk, 0, stream>>>(x, w0, b0, xxA);

    float* xx = xxA; float* xxN = xxB;

    for (int it = 0; it < 3; ++it) {
        float* edge = edgeb[it % 2];
        // ============ hgnn_layer: fg_mean ============
        // feat_nc[e,c] = sum_v H[v,e] * xx[c,v]   (TA: H, TB: xx)
        launch_gemm(true, true, Hbuf, xx, featN, NDIM, CDIM, NDIM,
                    NDIM, NDIM, CDIM, NNp, CNp, CNp, 1.f, 0, 0, stream);
        if (it == 0) {
            k_rowscale<<<ewg(CN), blk, 0, stream>>>(featN, deinv, edge, CDIM, 0);
        } else {
            float* ybuf = edgeb[(it + 1) % 2];           // previous layer's edge
            k_rowscale<<<ewg(CN), blk, 0, stream>>>(featN, deinv, featW, CDIM, 0);
            k_concat<<<ewg(NC2), blk, 0, stream>>>(ybuf, featW, catb);
            // mlp conv (C x 2C) on cat^T -> (C,N)
            launch_gemm(false, true, P(it, MLP_W), catb, Qb, CDIM, NDIM, 2 * CDIM,
                        2 * CDIM, 2 * CDIM, NDIM, 0, NC2p, CNp, 1.f, P(it, MLP_B), 0, stream);
            k_bnstats<<<dim3(CDIM), blk, 0, stream>>>(Qb, meanb, varb, CDIM);
            k_bnapply<<<ewg(CN), blk, 0, stream>>>(Qb, meanb, varb, P(it, MLP_G), P(it, MLP_BB),
                                                   0, Kb, CDIM, 1);
            k_rowscale<<<ewg(CN), blk, 0, stream>>>(Kb, 0, edge, CDIM, 1); // (C,N)->(N,C)
        }
        // featW = W_edge * edge ; feat2 = H @ featW ; feat2 *= dv_inv
        k_rowscale<<<ewg(CN), blk, 0, stream>>>(edge, wedge, featW, CDIM, 0);
        launch_gemm(false, false, Hbuf, featW, feat2, NDIM, CDIM, NDIM,
                    NDIM, CDIM, CDIM, NNp, CNp, CNp, 1.f, 0, 0, stream);
        k_rowscale<<<ewg(CN), blk, 0, stream>>>(feat2, dvinv, feat2, CDIM, 0);
        // x = lrelu(bnorm(con_w @ x1) + x)
        launch_gemm(false, true, P(it, CON_W), feat2, Qb, CDIM, NDIM, CDIM,
                    CDIM, CDIM, NDIM, 0, CNp, CNp, 1.f, 0, 0, stream);
        k_bnstats<<<dim3(CDIM), blk, 0, stream>>>(Qb, meanb, varb, CDIM);
        k_bnapply<<<ewg(CN), blk, 0, stream>>>(Qb, meanb, varb, P(it, BN_G), P(it, BN_B),
                                               xx, xxN, CDIM, 1);
        { float* t = xx; xx = xxN; xxN = t; }

        // ============ nonlocal_block ============
        launch_gemm(false, false, P(it, NL_QW), xx, Qb, CDIM, NDIM, CDIM,
                    CDIM, NDIM, NDIM, 0, CNp, CNp, 1.f, P(it, NL_QB), 0, stream);
        launch_gemm(false, false, P(it, NL_KW), xx, Kb, CDIM, NDIM, CDIM,
                    CDIM, NDIM, NDIM, 0, CNp, CNp, 1.f, P(it, NL_KB), 0, stream);
        launch_gemm(false, false, P(it, NL_VW), xx, Vb, CDIM, NDIM, CDIM,
                    CDIM, NDIM, NDIM, 0, CNp, CNp, 1.f, P(it, NL_VB), 0, stream);
        // fa = Q^T K / sqrt(C)
        launch_gemm(true, false, Qb, Kb, fabuf, NDIM, NDIM, CDIM,
                    NDIM, NDIM, NDIM, CNp, CNp, NNp, SCL, 0, 0, stream);
        const float* attw = (it == 0) ? Hbuf : Wmbuf;   // Wcur
        k_softmax<<<dim3(NDIM, 1, BATCH), blk, 0, stream>>>(fabuf, attw, Hbuf);
        // msg[c,o] = sum_i V[c,i] w[o,i]
        launch_gemm(false, true, Vb, fabuf, msg, CDIM, NDIM, NDIM,
                    NDIM, NDIM, NDIM, CNp, NNp, CNp, 1.f, 0, 0, stream);
        // fm1 -> relu(bn), fm2 -> relu(bn), fm3 + feat
        launch_gemm(false, false, P(it, FM1_W), msg, m1, CDIM / 2, NDIM, CDIM,
                    CDIM, NDIM, NDIM, 0, CNp, HNp, 1.f, P(it, FM1_B), 0, stream);
        k_bnstats<<<dim3(CDIM / 2), blk, 0, stream>>>(m1, meanb, varb, CDIM / 2);
        k_bnapply<<<ewg(HN), blk, 0, stream>>>(m1, meanb, varb, P(it, FM1_G), P(it, FM1_BB),
                                               0, m1, CDIM / 2, 2);
        launch_gemm(false, false, P(it, FM2_W), m1, m2, CDIM / 2, NDIM, CDIM / 2,
                    CDIM / 2, NDIM, NDIM, 0, HNp, HNp, 1.f, P(it, FM2_B), 0, stream);
        k_bnstats<<<dim3(CDIM / 2), blk, 0, stream>>>(m2, meanb, varb, CDIM / 2);
        k_bnapply<<<ewg(HN), blk, 0, stream>>>(m2, meanb, varb, P(it, FM2_G), P(it, FM2_BB),
                                               0, m2, CDIM / 2, 2);
        float* outp = (it == 2) ? (float*)d_out : xxN;
        launch_gemm(false, false, P(it, FM3_W), m2, outp, CDIM, NDIM, CDIM / 2,
                    CDIM / 2, NDIM, NDIM, 0, HNp, CNp, 1.f, P(it, FM3_B), xx, stream);

        if (it < 2) {
            { float* t = xx; xx = xxN; xxN = t; }
            // ============ graph_update ============
            launch_gemm(false, false, P(it, GU_QW), xx, Qb, CDIM, NDIM, CDIM,
                        CDIM, NDIM, NDIM, 0, CNp, CNp, 1.f, P(it, GU_QB), 0, stream);
            launch_gemm(false, true, P(it, GU_KW), edge, Kb, CDIM, NDIM, CDIM,
                        CDIM, CDIM, NDIM, 0, CNp, CNp, 1.f, P(it, GU_KB), 0, stream);
            launch_gemm(true, false, Qb, Kb, fabuf, NDIM, NDIM, CDIM,
                        NDIM, NDIM, NDIM, CNp, CNp, NNp, SCL, 0, 0, stream);
            k_sigmask<<<ewg(NN), blk, 0, stream>>>(fabuf, rawH);
            int kSel = (int)((double)NDIM * 0.1 * (double)(2 - it) + 0.5);  // 410, 205
            k_topk<<<dim3(NDIM, 1, BATCH), blk, 0, stream>>>(fabuf, rawH, Wmbuf, Hbuf, kSel);
            k_colsum<<<dim3((NDIM + 255) / 256, 1, BATCH), blk, 0, stream>>>(Hbuf, deg);
            k_inv<<<ewg(VN), blk, 0, stream>>>(deg, deinv, (int)VN);
            k_rowsum<<<dim3(NDIM, 1, BATCH), blk, 0, stream>>>(Hbuf, dvtmp);
            k_inv<<<ewg(VN), blk, 0, stream>>>(dvtmp, dvinv, (int)VN);
            k_colsum<<<dim3((NDIM + 255) / 256, 1, BATCH), blk, 0, stream>>>(Wmbuf, wsum);
            k_l2norm<<<dim3(BATCH), blk, 0, stream>>>(wsum, wedge);
        }
    }
}
